// GroupedQueryAttention_68667937128723
// MI455X (gfx1250) — compile-verified, matrix operand order checked by reading, not run on hardware
//
#include <hip/hip_runtime.h>

// Problem constants (from reference)
#define B_    2
#define S_    2048
#define D_    2048
#define H_    32
#define KVH_  8
#define HD_   64
#define HALF_ 32
#define REP_  4   // H / KVH

typedef __attribute__((ext_vector_type(16))) __bf16          v16bf;
typedef __attribute__((ext_vector_type(8)))  float           v8f;
typedef __attribute__((ext_vector_type(8)))  unsigned short  u16x8;
typedef __attribute__((ext_vector_type(4)))  unsigned int    u32x4;
typedef __attribute__((ext_vector_type(8)))  int             i32x8;
typedef __attribute__((ext_vector_type(4)))  int             i32x4;

// ---- Tensor Data Mover availability probes (compile-only safe) -------------
#if defined(__has_builtin)
# if __has_builtin(__builtin_amdgcn_tensor_load_to_lds) && \
     __has_builtin(__builtin_amdgcn_s_wait_tensorcnt)
#  define USE_TDM 1
# endif
#endif
#if defined(USE_TDM)
# if __has_include(<hip/amd_detail/amd_gfx1250_TDM.h>)
#  define TDM_ARGS6 1   // therock-10.0 headers -> 6-arg builtin
# endif
#endif

union BF16Frag {
    v16bf v;
    u16x8 h[2];
};

__device__ __forceinline__ unsigned short f2bf(float f) {
    unsigned u = __float_as_uint(f);
    u += 0x7FFFu + ((u >> 16) & 1u);   // round-to-nearest-even
    return (unsigned short)(u >> 16);
}

__device__ __forceinline__ v16bf load_frag(const unsigned short* p) {
    BF16Frag f;
    f.h[0] = *(const u16x8*)(p);
    f.h[1] = *(const u16x8*)(p + 16);
    return f.v;
}

__device__ __forceinline__ v8f wmma_bf16(v16bf a, v16bf b, v8f c) {
    return __builtin_amdgcn_wmma_f32_16x16x32_bf16(
        false, a, false, b, (short)0, c, false, false);
}

#if defined(USE_TDM)
// Issue one TDM 2D tile load: global (row-major, elem=2B) -> LDS, with LDS
// row padding 64B->80B (pad_interval=16 dwords, pad_amount=4 dwords) so the
// subsequent 16-row ds_load_b128 fragment reads are bank-conflict-free.
// Descriptor bitfields per CDNA5 ISA.
__device__ __forceinline__ void tdm_load_tile(unsigned lds_off,
                                              const void* gaddr,
                                              unsigned tensor_w,  // dim0 elems
                                              unsigned tensor_h,  // dim1 rows
                                              unsigned stride,    // dim0 stride
                                              unsigned tile_w,    // 32
                                              unsigned tile_h) {  // 128
    unsigned long long ga = (unsigned long long)gaddr;
    u32x4 g0;
    g0[0] = 1u;                                          // count=1 (user D#)
    g0[1] = lds_off;                                     // lds_addr (bytes)
    g0[2] = (unsigned)(ga & 0xFFFFFFFFu);                // global_addr lo
    g0[3] = (unsigned)((ga >> 32) & 0x1FFFFFFu) | (2u << 30);  // hi | type=2
    i32x8 g1;
    g1[0] = (1 << 16) | (1 << 20) | (3 << 22) | (3 << 25);
            // data_size=2B | pad_enable | pad_interval=16dw | pad_amount=4dw
    g1[1] = (int)((tensor_w & 0xFFFFu) << 16);           // dim0[15:0]
    g1[2] = (int)((tensor_w >> 16) | ((tensor_h & 0xFFFFu) << 16));
    g1[3] = (int)((tensor_h >> 16) | (tile_w << 16));    // dim1 hi | tile_dim0
    g1[4] = (int)(tile_h & 0xFFFFu);                     // tile_dim1 (tile_dim2=0)
    g1[5] = (int)stride;                                 // dim0_stride lo32
    g1[6] = 0;                                           // stride hi, dim1_stride
    g1[7] = 0;
    i32x4 z4 = {};
#if defined(TDM_ARGS6)
    i32x8 z8 = {};
    __builtin_amdgcn_tensor_load_to_lds(g0, g1, z4, z4, z8, 0);
#else
    __builtin_amdgcn_tensor_load_to_lds(g0, g1, z4, z4, 0);
#endif
}
#endif

// ---------------------------------------------------------------------------
// Elementwise conversion kernels (bandwidth-trivial vs. 23.3 TB/s HBM)
// ---------------------------------------------------------------------------
__global__ void cvt_bf16(const float* __restrict__ in,
                         unsigned short* __restrict__ out, long n) {
    long i = (long)blockIdx.x * blockDim.x + threadIdx.x;
    if (i < n) out[i] = f2bf(in[i]);
}

__global__ void cvt_bf16_t(const float* __restrict__ in,
                           unsigned short* __restrict__ out, int K, int N) {
    long i = (long)blockIdx.x * blockDim.x + threadIdx.x;
    if (i >= (long)K * N) return;
    int n = (int)(i % N);
    int k = (int)(i / N);
    out[(size_t)n * K + k] = f2bf(in[i]);
}

__global__ void rope_pack(const float* __restrict__ src,
                          const float* __restrict__ cs,
                          const float* __restrict__ sn,
                          unsigned short* __restrict__ dst, int Hn) {
    long i = (long)blockIdx.x * blockDim.x + threadIdx.x;
    long total = (long)B_ * S_ * Hn * HALF_;
    if (i >= total) return;
    int p = (int)(i % HALF_);
    int h = (int)((i / HALF_) % Hn);
    int s = (int)((i / ((long)HALF_ * Hn)) % S_);
    int b = (int)(i / ((long)HALF_ * Hn * S_));
    const float* r = src + (size_t)(b * S_ + s) * (Hn * HD_) + h * HD_ + 2 * p;
    float xr = r[0], xi = r[1];
    float c  = cs[(size_t)s * HALF_ + p];
    float si = sn[(size_t)s * HALF_ + p];
    unsigned short* d = dst + ((size_t)(b * Hn + h) * S_ + s) * HD_ + 2 * p;
    d[0] = f2bf(xr * c - xi * si);
    d[1] = f2bf(xr * si + xi * c);
}

__global__ void v_pack_t(const float* __restrict__ src,
                         unsigned short* __restrict__ dst) {
    long i = (long)blockIdx.x * blockDim.x + threadIdx.x;
    long total = (long)B_ * S_ * KVH_ * HD_;
    if (i >= total) return;
    int d = (int)(i % HD_);
    int h = (int)((i / HD_) % KVH_);
    int s = (int)((i / ((long)HD_ * KVH_)) % S_);
    int b = (int)(i / ((long)HD_ * KVH_ * S_));
    dst[((size_t)(b * KVH_ + h) * HD_ + d) * S_ + s] =
        f2bf(src[(size_t)(b * S_ + s) * (KVH_ * HD_) + h * HD_ + d]);
}

// ---------------------------------------------------------------------------
// bf16 WMMA GEMM with TDM-staged LDS tiles.
// C_f32[M,N] = A_bf16[M,K] @ Bt_bf16[N,K]^T.
// Block = 256 threads (8 waves), tile 128(M) x 128(N), k-step 32.
// Waves: 4(M) x 2(N); each wave owns 32x64 -> 8 chained WMMA per step.
// A/B 8KB tiles double-buffered in LDS (80B padded rows), filled by the
// Tensor Data Mover (TENSORcnt) so DMA overlaps WMMA; fallback is a
// cooperative b128 load / ds_store stage if the TDM builtin is absent.
// ---------------------------------------------------------------------------
#define RS_ 40   // LDS row stride in ushorts (80 bytes: 64B data + 16B pad)

__global__ __launch_bounds__(256)
void wmma_gemm_bf16(const unsigned short* __restrict__ A,
                    const unsigned short* __restrict__ Bt,
                    float* __restrict__ C, int M, int N, int K) {
    __shared__ unsigned short sA[2][128 * RS_];
    __shared__ unsigned short sB[2][128 * RS_];

    const int lane = threadIdx.x & 31;
    const int w    = threadIdx.x >> 5;
    const int wm   = w >> 1, wn = w & 1;
    const int hi   = lane >> 4, lo = lane & 15;
    const int m0   = blockIdx.x * 128;
    const int n0   = blockIdx.y * 128;

    v8f acc[2][4] = {};

#if defined(USE_TDM)
    const int nsteps = K / 32;
    const unsigned offA0 = (unsigned)(unsigned long long)(void*)&sA[0][0];
    const unsigned offA1 = (unsigned)(unsigned long long)(void*)&sA[1][0];
    const unsigned offB0 = (unsigned)(unsigned long long)(void*)&sB[0][0];
    const unsigned offB1 = (unsigned)(unsigned long long)(void*)&sB[1][0];
    if (w == 0) {
        tdm_load_tile(offA0, A  + (size_t)m0 * K, K, M - m0, K, 32, 128);
        tdm_load_tile(offB0, Bt + (size_t)n0 * K, K, N - n0, K, 32, 128);
    }
    for (int i = 0; i < nsteps; ++i) {
        const int buf = i & 1;
        if (w == 0) {
            if (i + 1 < nsteps) {
                int k1 = (i + 1) * 32;
                tdm_load_tile(buf ? offA0 : offA1,
                              A + (size_t)m0 * K + k1, K - k1, M - m0, K, 32, 128);
                tdm_load_tile(buf ? offB0 : offB1,
                              Bt + (size_t)n0 * K + k1, K - k1, N - n0, K, 32, 128);
                __builtin_amdgcn_s_wait_tensorcnt(2);  // 2 newest still in flight
            } else {
                __builtin_amdgcn_s_wait_tensorcnt(0);
            }
        }
        __syncthreads();
#else
    for (int i = 0; i < K / 32; ++i) {
        const int buf = 0;
        {   // cooperative stage: 256 threads x 32B cover each 8KB tile
            const int k0 = i * 32;
            const int row = threadIdx.x >> 1;
            const int hf  = (threadIdx.x & 1) * 16;
            const unsigned short* ga = A  + (size_t)(m0 + row) * K + k0 + hf;
            const unsigned short* gb = Bt + (size_t)(n0 + row) * K + k0 + hf;
            u16x8 a0 = *(const u16x8*)(ga);
            u16x8 a1 = *(const u16x8*)(ga + 8);
            u16x8 b0v = *(const u16x8*)(gb);
            u16x8 b1v = *(const u16x8*)(gb + 8);
            __builtin_prefetch(ga + 32, 0, 1);
            __builtin_prefetch(gb + 32, 0, 1);
            *(u16x8*)&sA[0][row * RS_ + hf]     = a0;
            *(u16x8*)&sA[0][row * RS_ + hf + 8] = a1;
            *(u16x8*)&sB[0][row * RS_ + hf]     = b0v;
            *(u16x8*)&sB[0][row * RS_ + hf + 8] = b1v;
        }
        __syncthreads();
#endif
        // ---- fragments from LDS + 8 WMMA
        BF16Frag a0f, a1f;
        const unsigned short* pa = &sA[buf][(wm * 32 + lo) * RS_ + hi * 8];
        a0f.h[0] = *(const u16x8*)(pa);
        a0f.h[1] = *(const u16x8*)(pa + 16);
        const unsigned short* pa1 = pa + 16 * RS_;
        a1f.h[0] = *(const u16x8*)(pa1);
        a1f.h[1] = *(const u16x8*)(pa1 + 16);
#pragma unroll
        for (int nt = 0; nt < 4; ++nt) {
            BF16Frag bfr;
            const unsigned short* pb =
                &sB[buf][(wn * 64 + nt * 16 + lo) * RS_ + hi * 8];
            bfr.h[0] = *(const u16x8*)(pb);
            bfr.h[1] = *(const u16x8*)(pb + 16);
            acc[0][nt] = wmma_bf16(a0f.v, bfr.v, acc[0][nt]);
            acc[1][nt] = wmma_bf16(a1f.v, bfr.v, acc[1][nt]);
        }
        __syncthreads();
    }

    // ---- epilogue: C layout (M = base + r + 8*hi, N = base + nt*16 + lo)
#pragma unroll
    for (int ms = 0; ms < 2; ++ms) {
        float* crow =
            C + (size_t)(m0 + wm * 32 + ms * 16 + 8 * hi) * N + n0 + wn * 64 + lo;
#pragma unroll
        for (int r = 0; r < 8; ++r) {
            crow[(size_t)r * N + 0]  = acc[ms][0][r];
            crow[(size_t)r * N + 16] = acc[ms][1][r];
            crow[(size_t)r * N + 32] = acc[ms][2][r];
            crow[(size_t)r * N + 48] = acc[ms][3][r];
        }
    }
}

// ---------------------------------------------------------------------------
// Flash-style causal GQA attention (unchanged from round 1; 8 WMMA per
// 32-key step, online softmax, P-tile C->A relayout through per-wave LDS).
// ---------------------------------------------------------------------------
__global__ __launch_bounds__(128)
void attn_fwd(const unsigned short* __restrict__ Qb,   // [B,H,S,HD]
              const unsigned short* __restrict__ Kb,   // [B,KVH,S,HD]
              const unsigned short* __restrict__ Vt,   // [B,KVH,HD,S]
              unsigned short* __restrict__ Ob) {       // [B,S,D]
    __shared__ unsigned short sP[4 * 16 * 40];

    const int lane = threadIdx.x & 31;
    const int w    = threadIdx.x >> 5;
    const int hi   = lane >> 4, lo = lane & 15;
    const int b    = blockIdx.z, h = blockIdx.y;
    const int kvh  = h / REP_;
    const int qw   = blockIdx.x * 64 + w * 16;

    const unsigned short* qbase =
        Qb + ((size_t)(b * H_ + h) * S_ + qw + lo) * HD_ + hi * 8;
    v16bf qf0 = load_frag(qbase);
    v16bf qf1 = load_frag(qbase + 32);

    const unsigned short* kbase =
        Kb + ((size_t)(b * KVH_ + kvh) * S_ + lo) * HD_ + hi * 8;
    const unsigned short* vbase =
        Vt + ((size_t)(b * KVH_ + kvh) * HD_ + lo) * S_ + hi * 8;

    v8f o0 = {}, o1 = {}, o2 = {}, o3 = {};
    float m[8], lsum[8];
#pragma unroll
    for (int r = 0; r < 8; ++r) { m[r] = -1e30f; lsum[r] = 0.0f; }

    unsigned short* Pw = &sP[w * 16 * 40];
    const float scale = 0.125f;

    for (int kk0 = 0; kk0 < qw + 16; kk0 += 32) {
        const unsigned short* kp = kbase + (size_t)kk0 * HD_;
        v16bf kf00 = load_frag(kp);
        v16bf kf01 = load_frag(kp + 32);
        v16bf kf10 = load_frag(kp + (size_t)16 * HD_);
        v16bf kf11 = load_frag(kp + (size_t)16 * HD_ + 32);
        v8f s0 = {}, s1 = {};
        s0 = wmma_bf16(qf0, kf00, s0);
        s0 = wmma_bf16(qf1, kf01, s0);
        s1 = wmma_bf16(qf0, kf10, s1);
        s1 = wmma_bf16(qf1, kf11, s1);

#pragma unroll
        for (int r = 0; r < 8; ++r) {
            int qi  = qw + r + 8 * hi;
            int ki0 = kk0 + lo;
            float a0 = (ki0      <= qi) ? s0[r] * scale : -1e30f;
            float a1 = (ki0 + 16 <= qi) ? s1[r] * scale : -1e30f;
            float mt = fmaxf(a0, a1);
            mt = fmaxf(mt, __shfl_xor(mt, 1));
            mt = fmaxf(mt, __shfl_xor(mt, 2));
            mt = fmaxf(mt, __shfl_xor(mt, 4));
            mt = fmaxf(mt, __shfl_xor(mt, 8));
            float mn    = fmaxf(m[r], mt);
            float alpha = __expf(m[r] - mn);
            m[r] = mn;
            float p0 = __expf(a0 - mn);
            float p1 = __expf(a1 - mn);
            float rs = p0 + p1;
            rs += __shfl_xor(rs, 1);
            rs += __shfl_xor(rs, 2);
            rs += __shfl_xor(rs, 4);
            rs += __shfl_xor(rs, 8);
            lsum[r] = lsum[r] * alpha + rs;
            o0[r] *= alpha; o1[r] *= alpha; o2[r] *= alpha; o3[r] *= alpha;
            int row = r + 8 * hi;
            Pw[row * 40 + lo]      = f2bf(p0);
            Pw[row * 40 + lo + 16] = f2bf(p1);
        }

        BF16Frag pf;
        const unsigned short* pp = &Pw[lo * 40 + hi * 8];
        pf.h[0] = *(const u16x8*)(pp);
        pf.h[1] = *(const u16x8*)(pp + 16);

        const unsigned short* vp = vbase + kk0;
        v16bf vf0 = load_frag(vp);
        v16bf vf1 = load_frag(vp + (size_t)16 * S_);
        v16bf vf2 = load_frag(vp + (size_t)32 * S_);
        v16bf vf3 = load_frag(vp + (size_t)48 * S_);
        o0 = wmma_bf16(pf.v, vf0, o0);
        o1 = wmma_bf16(pf.v, vf1, o1);
        o2 = wmma_bf16(pf.v, vf2, o2);
        o3 = wmma_bf16(pf.v, vf3, o3);
    }

    unsigned short* obase =
        Ob + ((size_t)b * S_ + qw + 8 * hi) * D_ + h * HD_ + lo;
#pragma unroll
    for (int r = 0; r < 8; ++r) {
        float inv = 1.0f / lsum[r];
        obase[(size_t)r * D_ + 0]  = f2bf(o0[r] * inv);
        obase[(size_t)r * D_ + 16] = f2bf(o1[r] * inv);
        obase[(size_t)r * D_ + 32] = f2bf(o2[r] * inv);
        obase[(size_t)r * D_ + 48] = f2bf(o3[r] * inv);
    }
}

// ---------------------------------------------------------------------------
extern "C" void kernel_launch(void* const* d_in, const int* in_sizes, int n_in,
                              void* d_out, int out_size, void* d_ws, size_t ws_size,
                              hipStream_t stream) {
    const float* x  = (const float*)d_in[0];
    const float* fc = (const float*)d_in[1];
    const float* fs = (const float*)d_in[2];
    const float* Wq = (const float*)d_in[3];
    const float* Wk = (const float*)d_in[4];
    const float* Wv = (const float*)d_in[5];
    const float* Wo = (const float*)d_in[6];
    float* out = (float*)d_out;
    char*  ws  = (char*)d_ws;

    unsigned short* xb  = (unsigned short*)(ws + 0);          // 16 MiB
    unsigned short* Wqb = (unsigned short*)(ws + 16777216);   //  8 MiB
    unsigned short* Wkb = (unsigned short*)(ws + 25165824);   //  2 MiB
    unsigned short* Wvb = (unsigned short*)(ws + 27262976);   //  2 MiB
    float*          Qf  = (float*)(ws + 29360128);            // 32 MiB
    float*          Kf  = (float*)(ws + 62914560);            //  8 MiB
    float*          Vf  = (float*)(ws + 71303168);            //  8 MiB
    unsigned short* Qb  = (unsigned short*)(ws + 79691776);   // 16 MiB
    unsigned short* Kb  = (unsigned short*)(ws + 96468992);   //  4 MiB
    unsigned short* Vt  = (unsigned short*)(ws + 100663296);  //  4 MiB
    unsigned short* Ob  = (unsigned short*)(ws + 29360128);   // alias Qf
    unsigned short* Wob = Wqb;                                // alias after Q GEMM

    const long nx = (long)B_ * S_ * D_;
    cvt_bf16<<<dim3((unsigned)((nx + 255) / 256)), 256, 0, stream>>>(x, xb, nx);
    cvt_bf16_t<<<dim3((unsigned)((2048L * 2048 + 255) / 256)), 256, 0, stream>>>(Wq, Wqb, 2048, 2048);
    cvt_bf16_t<<<dim3((unsigned)((2048L * 512 + 255) / 256)),  256, 0, stream>>>(Wk, Wkb, 2048, 512);
    cvt_bf16_t<<<dim3((unsigned)((2048L * 512 + 255) / 256)),  256, 0, stream>>>(Wv, Wvb, 2048, 512);

    wmma_gemm_bf16<<<dim3(32, 16), 256, 0, stream>>>(xb, Wqb, Qf, 4096, 2048, 2048);
    wmma_gemm_bf16<<<dim3(32, 4),  256, 0, stream>>>(xb, Wkb, Kf, 4096, 512, 2048);
    wmma_gemm_bf16<<<dim3(32, 4),  256, 0, stream>>>(xb, Wvb, Vf, 4096, 512, 2048);

    cvt_bf16_t<<<dim3((unsigned)((2048L * 2048 + 255) / 256)), 256, 0, stream>>>(Wo, Wob, 2048, 2048);

    rope_pack<<<dim3((unsigned)(((long)B_ * S_ * H_ * HALF_ + 255) / 256)),
                256, 0, stream>>>(Qf, fc, fs, Qb, H_);
    rope_pack<<<dim3((unsigned)(((long)B_ * S_ * KVH_ * HALF_ + 255) / 256)),
                256, 0, stream>>>(Kf, fc, fs, Kb, KVH_);
    v_pack_t<<<dim3((unsigned)(((long)B_ * S_ * KVH_ * HD_ + 255) / 256)),
               256, 0, stream>>>(Vf, Vt);

    attn_fwd<<<dim3(S_ / 64, H_, B_), 128, 0, stream>>>(Qb, Kb, Vt, Ob);

    wmma_gemm_bf16<<<dim3(32, 16), 256, 0, stream>>>(Ob, Wob, out, 4096, 2048, 2048);
}
